// KANLinear_71210557767745
// MI455X (gfx1250) — matrix operand order repaired and checked by reading, not compiled, not run on hardware
//
#include <hip/hip_runtime.h>
#include <hip/hip_bf16.h>

// ---------------------------------------------------------------------------
// KAN linear as one fused f16 WMMA GEMM on gfx1250:
//   A[8192 x 6144] (expanded: per input feature i -> [1{idx==0..4}, f16(x)])
//   W[1024 x 6144] (per i -> [sw[o,i,0..4], bw[o,i]] in f16)
//   C = A * W^T  (f32 accumulate)  ==  spline_out + base_out
// B tiles move via the Tensor Data Mover (iterate mode -> padded LDS pitch),
// double-buffered in LDS; A tiles are register-staged + VALU-expanded.
// ---------------------------------------------------------------------------

typedef _Float16 v16h __attribute__((ext_vector_type(16)));
typedef float    v8f  __attribute__((ext_vector_type(8)));
typedef unsigned v4u  __attribute__((ext_vector_type(4)));
typedef int      v8i  __attribute__((ext_vector_type(8)));
typedef int      v4i  __attribute__((ext_vector_type(4)));

#define BATCH   8192
#define IN_F    1024
#define OUT_F   1024
#define KTOT    (IN_F * 6)      // 6144 halves of K
#define BM      128
#define BN      128
#define BKH     96              // K-halves per tile (16 features * 6)
#define IGP     16              // input features per K tile
#define NKT     (IN_F / IGP)    // 64 K tiles
#define APITCH  104             // LDS row pitch in halves (96 + 8 pad)

#if __has_builtin(__builtin_amdgcn_tensor_load_to_lds)
#define USE_TDM 1
#else
#define USE_TDM 0
#endif

#define A_BYTES   (BM * APITCH * 2)
#define B_BYTES   (BN * APITCH * 2)
#if USE_TDM
#define LDS_BYTES (A_BYTES + 2 * B_BYTES)   // double-buffered B
#else
#define LDS_BYTES (A_BYTES + B_BYTES)
#endif

union Frag16 { v16h v; v4u q[2]; };

static __device__ __forceinline__ unsigned f16b(float v) {
    _Float16 h = (_Float16)v;
    unsigned short u;
    __builtin_memcpy(&u, &h, 2);
    return (unsigned)u;
}

#if USE_TDM
// TDM: copy one 128x96-half B tile, one row (192B) per iteration, stepping the
// LDS address by 104 halves per row to realize the bank-padded pitch.
static __device__ __forceinline__ void tdm_load_B(const _Float16* gptr,
                                                  unsigned lds_off) {
    unsigned long long ga = (unsigned long long)gptr;
    v4u g0;
    g0[0] = 1u;                                   // count=1, user desc, no gather
    g0[1] = lds_off;                              // lds_addr (bytes)
    g0[2] = (unsigned)ga;                         // global_addr[31:0]
    g0[3] = ((unsigned)(ga >> 32) & 0x01FFFFFFu)  // global_addr[56:32]
            | (2u << 30);                         // type = 2 ("image")
    v8i g1;
    g1[0] = (1 << 16) | (1 << 19);                // data_size=2B, iterate_enable
    g1[1] = (int)((unsigned)KTOT << 16);          // tensor_dim0[15:0] @ [63:48]
    g1[2] = (int)((unsigned)OUT_F << 16);         // dim0 hi=0; tensor_dim1[15:0]
    g1[3] = (int)((unsigned)BKH << 16);           // dim1 hi=0; tile_dim0=96
    g1[4] = 1;                                    // tile_dim1=1, tile_dim2=0
    g1[5] = KTOT;                                 // tensor_dim0_stride lo
    g1[6] = 0;                                    // stride hi / dim1_stride lo
    g1[7] = 0;
    v4i g2;
    g2[0] = 0;                                    // tensor_dim2 (unused)
    g2[1] = APITCH;                               // lds_addr_increment (elems)
    g2[2] = KTOT;                                 // global_addr_increment (elems)
    g2[3] = (int)((unsigned)(BN - 1) << 16);      // iterate_count: 128 rows
    v4i g3 = {0, 0, 0, 0};
    v8i g4 = {0, 0, 0, 0, 0, 0, 0, 0};            // unused trailing group (6-arg form)
    __builtin_amdgcn_tensor_load_to_lds(g0, g1, g2, g3, g4, 0);
}
#endif

// --- pre-pass 1: pack x -> u32 { idx[18:16] | f16(x)[15:0] } ----------------
__global__ void kan_pack_x(const float* __restrict__ x,
                           const float* __restrict__ grid,
                           unsigned* __restrict__ P, int n4) {
    int i = blockIdx.x * blockDim.x + threadIdx.x;
    if (i >= n4) return;
    const float g0 = grid[0], g1 = grid[1], g2 = grid[2];
    const float g3 = grid[3], g4 = grid[4], g5 = grid[5];
    float4 v = ((const float4*)x)[i];
    float vv[4] = {v.x, v.y, v.z, v.w};
    unsigned rr[4];
#pragma unroll
    for (int c = 0; c < 4; ++c) {
        float t = tanhf(vv[c]);
        unsigned idx = 7u;  // no interval -> all one-hots zero
        if (t >= g0 && t < g5)
            idx = (unsigned)((t >= g1) + (t >= g2) + (t >= g3) + (t >= g4));
        rr[c] = (idx << 16) | f16b(vv[c]);
    }
    uint4 r; r.x = rr[0]; r.y = rr[1]; r.z = rr[2]; r.w = rr[3];
    ((uint4*)P)[i] = r;
}

// --- pre-pass 2: pack weights -> f16 [o][i*6+c]: c<5 spline, c==5 base ------
__global__ void kan_pack_w(const float* __restrict__ sw,
                           const float* __restrict__ bw,
                           _Float16* __restrict__ Wh, int n) {
    int i = blockIdx.x * blockDim.x + threadIdx.x;   // flat (o,i) pair
    if (i >= n) return;
    const float* s = sw + (size_t)i * 8;             // 8 channels, use first 5
    float b = bw[i];
    unsigned* d = (unsigned*)(Wh + (size_t)i * 6);
    d[0] = f16b(s[0]) | (f16b(s[1]) << 16);
    d[1] = f16b(s[2]) | (f16b(s[3]) << 16);
    d[2] = f16b(s[4]) | (f16b(b)    << 16);
}

// --- main GEMM: C[8192x1024] = expand(P) * Wh^T  ----------------------------
__global__ __launch_bounds__(256, 2)
void kan_gemm(const unsigned* __restrict__ P,
              const _Float16* __restrict__ Wh,
              float* __restrict__ out) {
    extern __shared__ __align__(16) char smem[];
    _Float16* As = (_Float16*)smem;                 // [BM][APITCH]
    _Float16* Bs = (_Float16*)(smem + A_BYTES);     // [BN][APITCH] (x2 if TDM)

    const int tid  = threadIdx.x;
    const int lane = tid & 31;
    const int wave = tid >> 5;
    const int m0 = blockIdx.y * BM;
    const int n0 = blockIdx.x * BN;

    // ---- A loader partition ----
    const int a_iof = tid & 15;          // feature within K tile
    const int a_row = tid >> 4;          // 0..15; rows a_row + 16*j
    unsigned regA[8];

    auto loadA = [&](int t) {
        const unsigned* p = P + (size_t)(m0 + a_row) * IN_F + t * IGP + a_iof;
#pragma unroll
        for (int j = 0; j < 8; ++j) regA[j] = p[(size_t)j * 16 * IN_F];
    };
    auto storeA = [&]() {
#pragma unroll
        for (int j = 0; j < 8; ++j) {
            unsigned pk  = regA[j];
            unsigned idx = pk >> 16;
            unsigned xh  = pk & 0xffffu;
            const unsigned one = 0x3C00u;   // f16 1.0
            unsigned h0 = (idx == 0 ? one : 0u) | ((idx == 1 ? one : 0u) << 16);
            unsigned h1 = (idx == 2 ? one : 0u) | ((idx == 3 ? one : 0u) << 16);
            unsigned h2 = (idx == 4 ? one : 0u) | (xh << 16);
            unsigned* d = (unsigned*)(As + (a_row + 16 * j) * APITCH + a_iof * 6);
            d[0] = h0; d[1] = h1; d[2] = h2;
        }
    };

#if !USE_TDM
    const int b_n   = tid >> 1;          // 0..127 (output feature row)
    const int b_seg = (tid & 1) * 48;    // half of the 96-half row
    v4u regB[6];
    auto loadB = [&](int t) {
        const v4u* p = (const v4u*)(Wh + (size_t)(n0 + b_n) * KTOT + t * BKH + b_seg);
#pragma unroll
        for (int j = 0; j < 6; ++j) regB[j] = p[j];
    };
    auto storeB = [&]() {
#pragma unroll
        for (int j = 0; j < 6; ++j)
            *(v4u*)(Bs + b_n * APITCH + b_seg + j * 8) = regB[j];
    };
#endif

    // ---- wave tiling: 2x4 waves, each wave 64(M) x 32(N) = 4x2 WMMA tiles --
    const int wm  = (wave & 1) * 64;
    const int wn  = (wave >> 1) * 32;
    const int hs  = lane >> 4;           // half-wave select
    const int l16 = lane & 15;

    v8f acc[4][2] = {};

    // ---- prologue: stage tile 0 ----
#if USE_TDM
    const _Float16* Wtile = Wh + (size_t)n0 * KTOT;
    if (wave == 0)
        tdm_load_B(Wtile, (unsigned)(unsigned long long)(void*)Bs);
    loadA(0);
    storeA();
    if (wave == 0) __builtin_amdgcn_s_wait_tensorcnt(0);
#else
    loadA(0); loadB(0);
    storeA(); storeB();
#endif
    __syncthreads();

    for (int t = 0; t < NKT; ++t) {
#if USE_TDM
        _Float16* Bcur = Bs + (size_t)(t & 1) * BN * APITCH;
        if (t + 1 < NKT) {
            if (wave == 0) {
                _Float16* Bnxt = Bs + (size_t)((t + 1) & 1) * BN * APITCH;
                tdm_load_B(Wtile + (t + 1) * BKH,
                           (unsigned)(unsigned long long)(void*)Bnxt);
            }
            loadA(t + 1);
        }
#else
        _Float16* Bcur = Bs;
        if (t + 1 < NKT) { loadA(t + 1); loadB(t + 1); }
#endif

#pragma unroll
        for (int ks = 0; ks < 3; ++ks) {
            const int kk = ks * 32;
            Frag16 af[4], bf[2];
#pragma unroll
            for (int a = 0; a < 4; ++a) {
                // A 16x32 f16: lanes 0-15 K[kk..+7],[kk+16..+23];
                //              lanes 16-31 K[kk+8..+15],[kk+24..+31]
                const _Float16* base = As + (wm + a * 16 + l16) * APITCH + kk + hs * 8;
                af[a].q[0] = *(const v4u*)(base);
                af[a].q[1] = *(const v4u*)(base + 16);
            }
#pragma unroll
            for (int b = 0; b < 2; ++b) {
                // B 32x16 f16: lanes 0-15 K[kk..+7],[kk+16..+23]; lanes 16-31 K[kk+8..+15],[kk+24..+31]
                const _Float16* base = Bcur + (wn + b * 16 + l16) * APITCH + kk + hs * 8;
                bf[b].q[0] = *(const v4u*)(base);
                bf[b].q[1] = *(const v4u*)(base + 16);
            }
#pragma unroll
            for (int a = 0; a < 4; ++a)
#pragma unroll
                for (int b = 0; b < 2; ++b)
                    acc[a][b] = __builtin_amdgcn_wmma_f32_16x16x32_f16(
                        false, af[a].v, false, bf[b].v,
                        (short)0, acc[a][b], false, false);
        }

        __syncthreads();                 // all waves done reading A (and B tile)
        if (t + 1 < NKT) {
            storeA();                    // regs -> A buffer
#if USE_TDM
            if (wave == 0) __builtin_amdgcn_s_wait_tensorcnt(0);
#else
            storeB();
#endif
        }
        __syncthreads();                 // publish A(t+1) / B(t+1)
    }

    // ---- write C: VGPR r -> (M = r + 8*hs, N = l16) per 16x16 tile ---------
#pragma unroll
    for (int a = 0; a < 4; ++a) {
        const int row0 = m0 + wm + a * 16 + hs * 8;
#pragma unroll
        for (int b = 0; b < 2; ++b) {
            const int col = n0 + wn + b * 16 + l16;
            float* o = out + (size_t)row0 * OUT_F + col;
#pragma unroll
            for (int r = 0; r < 8; ++r)
                o[(size_t)r * OUT_F] = acc[a][b][r];
        }
    }
}

extern "C" void kernel_launch(void* const* d_in, const int* in_sizes, int n_in,
                              void* d_out, int out_size, void* d_ws, size_t ws_size,
                              hipStream_t stream) {
    const float* x    = (const float*)d_in[0];   // [8192,1024]
    const float* sw   = (const float*)d_in[1];   // [1024,1024,8]
    const float* bw   = (const float*)d_in[2];   // [1024,1024]
    const float* grid = (const float*)d_in[3];   // [6]
    float* out = (float*)d_out;

    // workspace layout: P (packed x, 32 MB) | Wh (f16 weights, 12 MB)
    unsigned*  P  = (unsigned*)d_ws;
    _Float16*  Wh = (_Float16*)((char*)d_ws + (size_t)BATCH * IN_F * 4);

    kan_pack_x<<<(BATCH * IN_F / 4 + 255) / 256, 256, 0, stream>>>(
        x, grid, P, BATCH * IN_F / 4);
    kan_pack_w<<<(OUT_F * IN_F + 255) / 256, 256, 0, stream>>>(
        sw, bw, Wh, OUT_F * IN_F);
    kan_gemm<<<dim3(OUT_F / BN, BATCH / BM), 256, LDS_BYTES, stream>>>(P, Wh, out);
}
